// BGNNA_33767032881163
// MI455X (gfx1250) — matrix operand order repaired and checked by reading, not compiled, not run on hardware
//
#include <hip/hip_runtime.h>
#include <hip/hip_bf16.h>

#define NN 12288
#define KSPLIT 4
#define KCHUNK (NN / KSPLIT) /* 3072 */

typedef __attribute__((ext_vector_type(16))) _Float16 v16h;
typedef __attribute__((ext_vector_type(8)))  float    v8f;
typedef __attribute__((ext_vector_type(4)))  float    v4f;

// ---------------------------------------------------------------------------
// Kernel 1: Bt[64][NN] fp16 = transpose([x@W | (x@W)^2]); W staged in LDS,
// tile transposed through LDS so global stores are coalesced 128B runs.
// ---------------------------------------------------------------------------
__global__ void __launch_bounds__(256) bgnn_prep(const float* __restrict__ x,
                                                 const float* __restrict__ w,
                                                 _Float16* __restrict__ Bt) {
  __shared__ float    wlds[64 * 32];
  __shared__ _Float16 btile[64 * 66];   // pad 66 to avoid bank conflicts
  const int tid = threadIdx.x;
  const int rowBase = blockIdx.x * 64;

  for (int i = tid; i < 64 * 32; i += 256) wlds[i] = w[i];
  __syncthreads();

  {
    const int c  = tid & 31;
    const int r0 = tid >> 5;
    #pragma unroll
    for (int jr = 0; jr < 8; ++jr) {
      const int r = jr * 8 + r0;
      const float* xr = x + (size_t)(rowBase + r) * 64;
      float acc = 0.f;
      #pragma unroll
      for (int kk = 0; kk < 64; ++kk) acc += xr[kk] * wlds[kk * 32 + c];
      btile[r * 66 + c]      = (_Float16)acc;
      btile[r * 66 + 32 + c] = (_Float16)(acc * acc);
    }
  }
  __syncthreads();
  {
    const int r  = tid & 63;
    const int c0 = tid >> 6;
    #pragma unroll
    for (int cc = 0; cc < 16; ++cc) {
      const int c = cc * 4 + c0;
      Bt[(size_t)c * NN + rowBase + r] = btile[r * 66 + c];
    }
  }
}

// ---------------------------------------------------------------------------
// Kernel 2: pure streaming GEMM  P = edge @ [xw | xw^2]  via f16 WMMA with
// f32 accumulate (identity / diagonal handled analytically in the epilogue).
// Each wave: 2 M-tiles (32 rows) x 4 N-tiles (64 cols), K split 4 ways.
// Also accumulates per-row sums of edge for the norm term.
// ---------------------------------------------------------------------------
__global__ void __launch_bounds__(128) bgnn_main(const float* __restrict__ edge,
                                                 const _Float16* __restrict__ Bt,
                                                 float* __restrict__ PS,
                                                 float* __restrict__ PT,
                                                 float* __restrict__ RS) {
  const int lane     = threadIdx.x & 31;
  // wave-uniform scalars via readfirstlane -> SALU loop control
  const int wid      = __builtin_amdgcn_readfirstlane(threadIdx.x >> 5);
  const int gw       = blockIdx.x * 4 + wid;
  const int split    = gw & (KSPLIT - 1);
  const int mg       = gw >> 2;            // 0..383  (32-row M group)
  const int kBegin   = split * KCHUNK;
  const int kEnd     = kBegin + KCHUNK;

  const int rowLocal = lane & 15;
  const int row0     = mg * 32 + rowLocal;
  const int koff     = (lane >> 4) * 16;   // A frag: cols k+koff+0..15
  const int kpart    = (lane >> 4) * 16;   // B frag: K 0..15 (lo lanes) / 16..31 (hi)

  const float*    aRow0 = edge + (size_t)row0 * NN + koff;
  const float*    aRow1 = aRow0 + (size_t)16 * NN;
  const _Float16* b0p   = Bt + (size_t)(0 * 16 + rowLocal) * NN + kpart;
  const _Float16* b1p   = Bt + (size_t)(1 * 16 + rowLocal) * NN + kpart;
  const _Float16* b2p   = Bt + (size_t)(2 * 16 + rowLocal) * NN + kpart;
  const _Float16* b3p   = Bt + (size_t)(3 * 16 + rowLocal) * NN + kpart;

  v8f c00 = {}, c01 = {}, c02 = {}, c03 = {};
  v8f c10 = {}, c11 = {}, c12 = {}, c13 = {};
  float rs0 = 0.f, rs1 = 0.f;

  for (int k = kBegin; k < kEnd; k += 32) {
    {   // keep both A streams ~8 chunks ahead (global_prefetch_b8, near cache)
      int kpf = k + 8 * 32;
      if (kpf > kEnd - 32) kpf = kEnd - 32;
      __builtin_prefetch((const void*)(aRow0 + kpf), 0, 3);
      __builtin_prefetch((const void*)(aRow1 + kpf), 0, 3);
    }
    v4f f0 = *(const v4f*)(aRow0 + k);
    v4f f1 = *(const v4f*)(aRow0 + k + 4);
    v4f f2 = *(const v4f*)(aRow0 + k + 8);
    v4f f3 = *(const v4f*)(aRow0 + k + 12);
    v4f g0 = *(const v4f*)(aRow1 + k);
    v4f g1 = *(const v4f*)(aRow1 + k + 4);
    v4f g2 = *(const v4f*)(aRow1 + k + 8);
    v4f g3 = *(const v4f*)(aRow1 + k + 12);

    v4f fsum = (f0 + f1) + (f2 + f3);
    rs0 += (fsum.x + fsum.y) + (fsum.z + fsum.w);
    v4f gsum = (g0 + g1) + (g2 + g3);
    rs1 += (gsum.x + gsum.y) + (gsum.z + gsum.w);

    v16h a0, a1;
    #pragma unroll
    for (int j = 0; j < 4; ++j) {
      a0[j]      = (_Float16)f0[j];
      a0[4 + j]  = (_Float16)f1[j];
      a0[8 + j]  = (_Float16)f2[j];
      a0[12 + j] = (_Float16)f3[j];
      a1[j]      = (_Float16)g0[j];
      a1[4 + j]  = (_Float16)g1[j];
      a1[8 + j]  = (_Float16)g2[j];
      a1[12 + j] = (_Float16)g3[j];
    }

    v16h b0 = *(const v16h*)(b0p + k);
    v16h b1 = *(const v16h*)(b1p + k);
    v16h b2 = *(const v16h*)(b2p + k);
    v16h b3 = *(const v16h*)(b3p + k);

    c00 = __builtin_amdgcn_wmma_f32_16x16x32_f16(false, a0, false, b0, (short)0, c00, false, false);
    c10 = __builtin_amdgcn_wmma_f32_16x16x32_f16(false, a1, false, b0, (short)0, c10, false, false);
    c01 = __builtin_amdgcn_wmma_f32_16x16x32_f16(false, a0, false, b1, (short)0, c01, false, false);
    c11 = __builtin_amdgcn_wmma_f32_16x16x32_f16(false, a1, false, b1, (short)0, c11, false, false);
    c02 = __builtin_amdgcn_wmma_f32_16x16x32_f16(false, a0, false, b2, (short)0, c02, false, false);
    c12 = __builtin_amdgcn_wmma_f32_16x16x32_f16(false, a1, false, b2, (short)0, c12, false, false);
    c03 = __builtin_amdgcn_wmma_f32_16x16x32_f16(false, a0, false, b3, (short)0, c03, false, false);
    c13 = __builtin_amdgcn_wmma_f32_16x16x32_f16(false, a1, false, b3, (short)0, c13, false, false);
  }

  // C/D layout: lanes 0-15 hold M=v, lanes 16-31 hold M=v+8; N = lane & 15.
  const int half  = (lane < 16) ? 0 : 8;
  #pragma unroll
  for (int m = 0; m < 2; ++m) {
    const int rbase = mg * 32 + m * 16 + half;
    float* ps = PS + (size_t)(split * NN + rbase) * 32 + rowLocal;
    float* pt = PT + (size_t)(split * NN + rbase) * 32 + rowLocal;
    const v8f s0 = m ? c10 : c00;
    const v8f s1 = m ? c11 : c01;
    const v8f t0 = m ? c12 : c02;
    const v8f t1 = m ? c13 : c03;
    #pragma unroll
    for (int v = 0; v < 8; ++v) {
      ps[(size_t)v * 32]      = s0[v];
      ps[(size_t)v * 32 + 16] = s1[v];
      pt[(size_t)v * 32]      = t0[v];
      pt[(size_t)v * 32 + 16] = t1[v];
    }
  }
  RS[(size_t)(split * 2 + (lane >> 4)) * NN + mg * 32 + rowLocal]      = rs0;
  RS[(size_t)(split * 2 + (lane >> 4)) * NN + mg * 32 + 16 + rowLocal] = rs1;
}

// ---------------------------------------------------------------------------
// Kernel 3: reduce K-splits, add the identity (diagonal) analytically, apply
// norm / bias.  With e = edge[i,i], a = e+1 (diag of adj):
//   S_adj       = S_edge + xw_h[i,c]
//   A^2 @ xw^2  = T_edge + (a^2 - e) * xw2_h[i,c]
//   rowsum_adj  = rs + 1 ;  sum(adj^2) = rs - e + a^2
// ---------------------------------------------------------------------------
__global__ void __launch_bounds__(256) bgnn_epilogue(const float* __restrict__ edge,
                                                     const _Float16* __restrict__ Bt,
                                                     const float* __restrict__ PS,
                                                     const float* __restrict__ PT,
                                                     const float* __restrict__ RS,
                                                     const float* __restrict__ bias,
                                                     float* __restrict__ out) {
  const int idx = blockIdx.x * 256 + threadIdx.x;
  if (idx >= NN * 32) return;
  const int i = idx >> 5;
  const int c = idx & 31;

  float s = 0.f, t = 0.f;
  #pragma unroll
  for (int q = 0; q < KSPLIT; ++q) {
    s += PS[(size_t)(q * NN + i) * 32 + c];
    t += PT[(size_t)(q * NN + i) * 32 + c];
  }
  float rs = 0.f;
  #pragma unroll
  for (int q = 0; q < 2 * KSPLIT; ++q) rs += RS[(size_t)q * NN + i];

  const float e  = edge[(size_t)i * NN + i];
  const float a  = e + 1.0f;
  const float rsum  = rs + 1.0f;                 // rowsum of adj
  const float sumsq = rs - e + a * a;            // sum of adj^2 over the row
  const float denom = rsum * rsum - sumsq;
  float nrm = (denom != 0.0f) ? (1.0f / denom) : 0.0f;
  if (__builtin_isinf(nrm)) nrm = 0.0f;

  const float xwh  = (float)Bt[(size_t)c * NN + i];
  const float xw2h = (float)Bt[(size_t)(32 + c) * NN + i];
  const float S  = s + xwh;                      // adj @ xw
  const float T2 = t + (a * a - e) * xw2h;       // adj^2 @ xw^2
  out[idx] = nrm * (S * S - T2) + bias[c];
}

// ---------------------------------------------------------------------------
extern "C" void kernel_launch(void* const* d_in, const int* in_sizes, int n_in,
                              void* d_out, int out_size, void* d_ws, size_t ws_size,
                              hipStream_t stream) {
  (void)in_sizes; (void)n_in; (void)out_size; (void)ws_size;
  const float* x    = (const float*)d_in[0];
  const float* edge = (const float*)d_in[1];
  const float* w    = (const float*)d_in[2];
  const float* bias = (const float*)d_in[3];
  float* out = (float*)d_out;

  char* ws = (char*)d_ws;
  size_t off = 0;
  _Float16* Bt = (_Float16*)(ws + off);
  off += ((size_t)64 * NN * sizeof(_Float16) + 255) & ~(size_t)255;       // 1.5 MB
  float* PS = (float*)(ws + off);
  off += ((size_t)KSPLIT * NN * 32 * sizeof(float) + 255) & ~(size_t)255; // 6.3 MB
  float* PT = (float*)(ws + off);
  off += ((size_t)KSPLIT * NN * 32 * sizeof(float) + 255) & ~(size_t)255; // 6.3 MB
  float* RS = (float*)(ws + off);                                          // 0.4 MB

  bgnn_prep    <<<NN / 64,                 256, 0, stream>>>(x, w, Bt);
  bgnn_main    <<<(NN / 32) * KSPLIT / 4,  128, 0, stream>>>(edge, Bt, PS, PT, RS);
  bgnn_epilogue<<<(NN * 32) / 256,         256, 0, stream>>>(edge, Bt, PS, PT, RS, bias, out);
}
